// CTNet_30279519436809
// MI455X (gfx1250) — compile-verified, matrix operand order checked by reading, not run on hardware
//
#include <hip/hip_runtime.h>
#include <math.h>

typedef __attribute__((ext_vector_type(16))) _Float16 v16h;
typedef __attribute__((ext_vector_type(8)))  float    v8f;

#define Bc    16
#define Nn    1024
#define NBc   16384
#define Ec    262144
#define CINc  64
#define HIDc  32
#define K1c   100
#define K1P   112
#define K2c   16
#define COUTc 10
#define EPSc  1e-15f

// ---------------- workspace layout (float offsets) ----------------
// Zero-initialized accumulation block first (contiguous):
#define OFF_AGG   0                           // NB*HID   (adj2 @ h1)
#define OFF_T2    (OFF_AGG + NBc*HIDc)        // NB*K2    (adj2 @ s2)
#define OFF_ODEG  (OFF_T2 + NBc*K2c)          // NB       (adj row sums, counts)
#define OFF_DEG2  (OFF_ODEG + NBc)            // NB       (raw CT-dist row sums)
#define OFF_DEGA2 (OFF_DEG2 + NBc)            // NB       (normalized adj2 row sums)
#define OFF_SCAL  (OFF_DEGA2 + NBc)           // 128      (per-b scalars)
#define ZTOT      (OFF_SCAL + 128)
// Overwritten scratch:
#define OFF_H1    ZTOT                        // NB*HID
#define OFF_H2    (OFF_H1 + NBc*HIDc)         // NB*HID
#define OFF_S1    (OFF_H2 + NBc*HIDc)         // NB*K1P (cols >=100 are zero)
#define OFF_S2    (OFF_S1 + NBc*K1P)          // NB*K2
#define OFF_Q1    (OFF_S2 + NBc*K2c)          // NB  ||s1_n||^2
#define OFF_Q2    (OFF_Q1 + NBc)              // NB  ||s2_n||^2
#define OFF_EDOT  (OFF_Q2 + NBc)              // E   s1[src].s1[dst]
#define OFF_EW    (OFF_EDOT + Ec)             // E   edge weight (raw then normalized)
#define OFF_SS1   (OFF_EW + Ec)               // B*K1P*K1P
#define OFF_SS2   (OFF_SS1 + Bc*K1P*K1P)      // B*16*16
#define OFF_OUTP  (OFF_SS2 + Bc*K2c*K2c)      // B*16*32
#define OFF_OADJ  (OFF_OUTP + Bc*K2c*HIDc)    // B*16*16
#define OFF_H3    (OFF_OADJ + Bc*K2c*K2c)     // B*16*32
#define OFF_HS    (OFF_H3 + Bc*K2c*HIDc)      // B*32

// per-b scalar slots inside SCAL
#define SC_SDS  0
#define SC_SAS  16
#define SC_O1   32
#define SC_NUM  48
#define SC_DEN  64
#define SC_O2   80

// -------- WMMA fragment helpers (ISA 7.12.2 16-bit layouts, wave32) --------
// A (16xK f16): lane holds row m=lane&15; element e -> K index:
//   v=e>>1: base = v<4 ? 2v : 16+2(v-4);  K = base + (e&1) + 8*(lane>>4)
// B (Kx16 f16): same K mapping, lane holds column n=lane&15.
__device__ __forceinline__ v16h build_frag(const _Float16* p, int sK, int sM) {
  const int lane = threadIdx.x & 31;
  const int m = lane & 15, half = lane >> 4;
  v16h f;
#pragma unroll
  for (int e = 0; e < 16; ++e) {
    const int v = e >> 1, lo = e & 1;
    const int k = ((v < 4) ? (v << 1) : (16 + ((v - 4) << 1))) + lo + (half << 3);
    f[e] = p[k * sK + m * sM];
  }
  return f;
}

__device__ __forceinline__ v8f wmma16(v16h a, v16h b, v8f c) {
  return __builtin_amdgcn_wmma_f32_16x16x32_f16(false, a, false, b, (short)0, c,
                                                false, false);
}
// C/D: acc[j] -> row (j + 8*(lane>>4)), col (lane&15)

// ---------------------------- kernels ----------------------------
__global__ void k_zero(float* __restrict__ p, int n) {
  int i = blockIdx.x * blockDim.x + threadIdx.x;
  if (i < n) p[i] = 0.0f;
}

// h1 = x @ lin1_w + lin1_b   [NB,64]x[64,32]
__global__ __launch_bounds__(64) void k_lin1(const float* __restrict__ x,
                                             const float* __restrict__ w,
                                             const float* __restrict__ bias,
                                             float* __restrict__ h1) {
  __shared__ _Float16 A[16 * CINc];
  __shared__ _Float16 Bm[CINc * HIDc];
  const int tid = threadIdx.x, m0 = blockIdx.x * 16;
  for (int i = tid; i < 16 * CINc; i += 64) A[i] = (_Float16)x[m0 * CINc + i];
  for (int i = tid; i < CINc * HIDc; i += 64) Bm[i] = (_Float16)w[i];
  __syncthreads();
  const int wv = tid >> 5;
  v8f acc = {};
#pragma unroll
  for (int kk = 0; kk < CINc; kk += 32) {
    v16h a = build_frag(A + kk, 1, CINc);
    v16h b = build_frag(Bm + kk * HIDc + wv * 16, HIDc, 1);
    acc = wmma16(a, b, acc);
  }
  const int lane = tid & 31, col = lane & 15, half = lane >> 4;
#pragma unroll
  for (int j = 0; j < 8; ++j) {
    int r = j + 8 * half, c = wv * 16 + col;
    h1[(m0 + r) * HIDc + c] = acc[j] + bias[c];
  }
}

// s1 = h1 @ pool1_w + pool1_b, padded to 112 cols (pad cols exactly zero)
__global__ __launch_bounds__(256) void k_pool1(const float* __restrict__ h1,
                                               const float* __restrict__ w,
                                               const float* __restrict__ bias,
                                               float* __restrict__ s1) {
  __shared__ _Float16 A[16 * HIDc];
  __shared__ _Float16 Bm[HIDc * K1P];
  const int tid = threadIdx.x, m0 = blockIdx.x * 16;
  for (int i = tid; i < 16 * HIDc; i += 256) A[i] = (_Float16)h1[m0 * HIDc + i];
  for (int i = tid; i < HIDc * K1P; i += 256) {
    int k = i / K1P, c = i % K1P;
    Bm[i] = (c < K1c) ? (_Float16)w[k * K1c + c] : (_Float16)0.0f;
  }
  __syncthreads();
  const int wv = tid >> 5;
  if (wv < 7) {
    v16h a = build_frag(A, 1, HIDc);
    v16h b = build_frag(Bm + wv * 16, K1P, 1);
    v8f acc = {};
    acc = wmma16(a, b, acc);
    const int lane = tid & 31, col = lane & 15, half = lane >> 4;
#pragma unroll
    for (int j = 0; j < 8; ++j) {
      int r = j + 8 * half, c = wv * 16 + col;
      s1[(size_t)(m0 + r) * K1P + c] = acc[j] + ((c < K1c) ? bias[c] : 0.0f);
    }
  }
}

// softmax over first 100 cols, in place; q1 = sum(p^2)
__global__ void k_softmax1(float* __restrict__ s1, float* __restrict__ q1) {
  int n = blockIdx.x * blockDim.x + threadIdx.x;
  if (n >= NBc) return;
  float* row = s1 + (size_t)n * K1P;
  float mx = -1e30f;
  for (int k = 0; k < K1c; ++k) mx = fmaxf(mx, row[k]);
  float sum = 0.f;
  for (int k = 0; k < K1c; ++k) { float e = expf(row[k] - mx); row[k] = e; sum += e; }
  float inv = 1.f / sum, q = 0.f;
  for (int k = 0; k < K1c; ++k) { float p = row[k] * inv; row[k] = p; q += p * p; }
  q1[n] = q;
}

__device__ __forceinline__ void edge_nodes(const int* ei, int e, int& b, int& sg, int& dg) {
  int src = ei[e], dst = ei[Ec + e];
  b = src / Nn;
  sg = b * Nn + (src % Nn);
  dg = b * Nn + (dst % Nn);
}

// per edge: dot(s1[src],s1[dst]) -> edot; sas[b] += dot; odeg[src] += 1
__global__ __launch_bounds__(256) void k_edges1(const int* __restrict__ ei,
                                                const float* __restrict__ s1,
                                                float* __restrict__ edot,
                                                float* __restrict__ odeg,
                                                float* __restrict__ sas) {
  __shared__ float sacc[Bc];
  int tid = threadIdx.x, e = blockIdx.x * 256 + tid;
  if (tid < Bc) sacc[tid] = 0.f;
  __syncthreads();
  int b, sg, dg;
  edge_nodes(ei, e, b, sg, dg);
  const float* a = s1 + (size_t)sg * K1P;
  const float* c = s1 + (size_t)dg * K1P;
  float dot = 0.f;
  for (int k = 0; k < K1c; ++k) dot += a[k] * c[k];
  edot[e] = dot;
  atomicAdd(&sacc[b], dot);
  atomicAdd(&odeg[sg], 1.0f);
  __syncthreads();
  if (tid < Bc) atomicAdd(&sas[tid], sacc[tid]);
}

// sds[b] += odeg[n] * q1[n]
__global__ __launch_bounds__(256) void k_sds(const float* __restrict__ odeg,
                                             const float* __restrict__ q1,
                                             float* __restrict__ sds) {
  __shared__ float sacc[Bc];
  int tid = threadIdx.x, n = blockIdx.x * 256 + tid;
  if (tid < Bc) sacc[tid] = 0.f;
  __syncthreads();
  atomicAdd(&sacc[n / Nn], odeg[n] * q1[n]);
  __syncthreads();
  if (tid < Bc) atomicAdd(&sds[tid], sacc[tid]);
}

// raw CT edge weight; deg2[src] += w
__global__ void k_ew1(const int* __restrict__ ei, const float* __restrict__ q1,
                      const float* __restrict__ edot, float* __restrict__ ew,
                      float* __restrict__ deg2) {
  int e = blockIdx.x * blockDim.x + threadIdx.x;
  if (e >= Ec) return;
  int b, sg, dg;
  edge_nodes(ei, e, b, sg, dg);
  float d2 = q1[sg] + q1[dg] - 2.0f * edot[e];
  float w = (d2 > 0.f) ? sqrtf(d2) : 0.f;
  ew[e] = w;
  atomicAdd(&deg2[sg], w);
}

// symmetric degree-normalize edge weight; degA2[src] += w_norm
__global__ void k_ew2(const int* __restrict__ ei, const float* __restrict__ deg2,
                      float* __restrict__ ew, float* __restrict__ degA2) {
  int e = blockIdx.x * blockDim.x + threadIdx.x;
  if (e >= Ec) return;
  int b, sg, dg;
  edge_nodes(ei, e, b, sg, dg);
  float wn = ew[e] / ((sqrtf(deg2[sg]) + EPSc) * (sqrtf(deg2[dg]) + EPSc));
  ew[e] = wn;
  atomicAdd(&degA2[sg], wn);
}

// agg[src,:] += w * h1[dst,:]   (adj2 @ h1)
__global__ void k_scatter_h(const int* __restrict__ ei, const float* __restrict__ ew,
                            const float* __restrict__ h1, float* __restrict__ agg) {
  int idx = blockIdx.x * blockDim.x + threadIdx.x;
  int e = idx >> 5, k = idx & 31;
  int b, sg, dg;
  edge_nodes(ei, e, b, sg, dg);
  atomicAdd(&agg[sg * HIDc + k], ew[e] * h1[dg * HIDc + k]);
}

// h2 = agg @ rel_w + rel_b + h1 @ root_w  (two chained WMMAs)
__global__ __launch_bounds__(64) void k_conv1(const float* __restrict__ agg,
                                              const float* __restrict__ h1,
                                              const float* __restrict__ relw,
                                              const float* __restrict__ relb,
                                              const float* __restrict__ rootw,
                                              float* __restrict__ h2) {
  __shared__ _Float16 Ag[16 * HIDc], Ah[16 * HIDc];
  __shared__ _Float16 Wr[HIDc * HIDc], Wo[HIDc * HIDc];
  const int tid = threadIdx.x, m0 = blockIdx.x * 16;
  for (int i = tid; i < 16 * HIDc; i += 64) {
    Ag[i] = (_Float16)agg[m0 * HIDc + i];
    Ah[i] = (_Float16)h1[m0 * HIDc + i];
  }
  for (int i = tid; i < HIDc * HIDc; i += 64) {
    Wr[i] = (_Float16)relw[i];
    Wo[i] = (_Float16)rootw[i];
  }
  __syncthreads();
  const int wv = tid >> 5;
  v8f acc = {};
  acc = wmma16(build_frag(Ag, 1, HIDc), build_frag(Wr + wv * 16, HIDc, 1), acc);
  acc = wmma16(build_frag(Ah, 1, HIDc), build_frag(Wo + wv * 16, HIDc, 1), acc);
  const int lane = tid & 31, col = lane & 15, half = lane >> 4;
#pragma unroll
  for (int j = 0; j < 8; ++j) {
    int r = j + 8 * half, c = wv * 16 + col;
    h2[(m0 + r) * HIDc + c] = acc[j] + relb[c];
  }
}

// s2 = h2 @ pool2_w + pool2_b
__global__ __launch_bounds__(32) void k_pool2(const float* __restrict__ h2,
                                              const float* __restrict__ w,
                                              const float* __restrict__ bias,
                                              float* __restrict__ s2) {
  __shared__ _Float16 A[16 * HIDc], Bm[HIDc * K2c];
  const int tid = threadIdx.x, m0 = blockIdx.x * 16;
  for (int i = tid; i < 16 * HIDc; i += 32) A[i] = (_Float16)h2[m0 * HIDc + i];
  for (int i = tid; i < HIDc * K2c; i += 32) Bm[i] = (_Float16)w[i];
  __syncthreads();
  v8f acc = {};
  acc = wmma16(build_frag(A, 1, HIDc), build_frag(Bm, K2c, 1), acc);
  const int col = tid & 15, half = tid >> 4;
#pragma unroll
  for (int j = 0; j < 8; ++j) {
    int r = j + 8 * half;
    s2[(m0 + r) * K2c + col] = acc[j] + bias[col];
  }
}

__global__ void k_softmax2(float* __restrict__ s2, float* __restrict__ q2) {
  int n = blockIdx.x * blockDim.x + threadIdx.x;
  if (n >= NBc) return;
  float* row = s2 + (size_t)n * K2c;
  float mx = -1e30f;
  for (int k = 0; k < K2c; ++k) mx = fmaxf(mx, row[k]);
  float sum = 0.f;
  for (int k = 0; k < K2c; ++k) { float e = expf(row[k] - mx); row[k] = e; sum += e; }
  float inv = 1.f / sum, q = 0.f;
  for (int k = 0; k < K2c; ++k) { float p = row[k] * inv; row[k] = p; q += p * p; }
  q2[n] = q;
}

// den[b] += degA2[n] * q2[n]
__global__ __launch_bounds__(256) void k_den2(const float* __restrict__ degA2,
                                              const float* __restrict__ q2,
                                              float* __restrict__ den) {
  __shared__ float sacc[Bc];
  int tid = threadIdx.x, n = blockIdx.x * 256 + tid;
  if (tid < Bc) sacc[tid] = 0.f;
  __syncthreads();
  atomicAdd(&sacc[n / Nn], degA2[n] * q2[n]);
  __syncthreads();
  if (tid < Bc) atomicAdd(&den[tid], sacc[tid]);
}

// t2[src,:] += w * s2[dst,:]   (adj2 @ s2)
__global__ void k_scatter_t2(const int* __restrict__ ei, const float* __restrict__ ew,
                             const float* __restrict__ s2, float* __restrict__ t2) {
  int idx = blockIdx.x * blockDim.x + threadIdx.x;
  int e = idx >> 4, k = idx & 15;
  int b, sg, dg;
  edge_nodes(ei, e, b, sg, dg);
  atomicAdd(&t2[sg * K2c + k], ew[e] * s2[dg * K2c + k]);
}

// SS1[b] = S1^T S1 (112x112, K=1024), one 16x16 tile per block
__global__ __launch_bounds__(32) void k_ss1(const float* __restrict__ s1,
                                            float* __restrict__ ss1) {
  __shared__ _Float16 At[32 * 16], Bt[32 * 16];
  const int b = blockIdx.x / 49, t = blockIdx.x % 49, mi = t / 7, ni = t % 7;
  const int tid = threadIdx.x;
  v8f acc = {};
  for (int kk = 0; kk < Nn; kk += 32) {
    for (int i = tid; i < 512; i += 32) {
      int n = kk + i / 16, c = i % 16;
      const float* base = s1 + ((size_t)b * Nn + n) * K1P;
      At[i] = (_Float16)base[mi * 16 + c];
      Bt[i] = (_Float16)base[ni * 16 + c];
    }
    __syncthreads();
    acc = wmma16(build_frag(At, 16, 1), build_frag(Bt, 16, 1), acc);
    __syncthreads();
  }
  const int col = tid & 15, half = tid >> 4;
#pragma unroll
  for (int j = 0; j < 8; ++j) {
    int r = j + 8 * half;
    ss1[(size_t)b * K1P * K1P + (size_t)(mi * 16 + r) * K1P + ni * 16 + col] = acc[j];
  }
}

// per graph: outp = S2^T h2 ; oadj = S2^T t2 ; ss2 = S2^T S2
__global__ __launch_bounds__(64) void k_bgemms(const float* __restrict__ s2,
                                               const float* __restrict__ h2,
                                               const float* __restrict__ t2,
                                               float* __restrict__ outp,
                                               float* __restrict__ oadj,
                                               float* __restrict__ ss2) {
  __shared__ _Float16 St[32 * K2c], Ht[32 * HIDc], Tt[32 * K2c];
  const int tid = threadIdx.x, b = blockIdx.x, wv = tid >> 5;
  v8f acc0 = {}, acc1 = {};
  for (int kk = 0; kk < Nn; kk += 32) {
    for (int i = tid; i < 32 * K2c; i += 64) {
      int n = kk + i / K2c, c = i % K2c;
      St[i] = (_Float16)s2[((size_t)b * Nn + n) * K2c + c];
      Tt[i] = (_Float16)t2[((size_t)b * Nn + n) * K2c + c];
    }
    for (int i = tid; i < 32 * HIDc; i += 64) {
      int n = kk + i / HIDc, c = i % HIDc;
      Ht[i] = (_Float16)h2[((size_t)b * Nn + n) * HIDc + c];
    }
    __syncthreads();
    v16h a = build_frag(St, K2c, 1);  // A[m,k] = s2[k][m]  (S^T)
    if (wv == 0) {
      acc0 = wmma16(a, build_frag(Ht, HIDc, 1), acc0);
      acc1 = wmma16(a, build_frag(Tt, K2c, 1), acc1);
    } else {
      acc0 = wmma16(a, build_frag(Ht + 16, HIDc, 1), acc0);
      acc1 = wmma16(a, build_frag(St, K2c, 1), acc1);
    }
    __syncthreads();
  }
  const int lane = tid & 31, col = lane & 15, half = lane >> 4;
#pragma unroll
  for (int j = 0; j < 8; ++j) {
    int r = j + 8 * half;
    if (wv == 0) {
      outp[(b * K2c + r) * HIDc + col] = acc0[j];
      oadj[(b * K2c + r) * K2c + col] = acc1[j];
    } else {
      outp[(b * K2c + r) * HIDc + 16 + col] = acc0[j];
      ss2[(b * K2c + r) * K2c + col] = acc1[j];
    }
  }
}

// ortho loss term per graph: || SS/||SS||_F - I*diagval ||_F
__global__ __launch_bounds__(256) void k_ortho(const float* __restrict__ ss, int dim,
                                               int kdiag, float diagval,
                                               float* __restrict__ out) {
  __shared__ float red[256];
  __shared__ float s_norm;
  const int b = blockIdx.x, tid = threadIdx.x, tot = dim * dim;
  const float* p = ss + (size_t)b * tot;
  float sum = 0.f;
  for (int i = tid; i < tot; i += 256) { float v = p[i]; sum += v * v; }
  red[tid] = sum;
  __syncthreads();
  for (int s = 128; s > 0; s >>= 1) { if (tid < s) red[tid] += red[tid + s]; __syncthreads(); }
  if (tid == 0) s_norm = sqrtf(fmaxf(red[0], 1e-30f));
  __syncthreads();
  const float inv = 1.f / s_norm;
  sum = 0.f;
  for (int i = tid; i < tot; i += 256) {
    int r = i / dim, c = i % dim;
    float v = p[i] * inv - ((r == c && r < kdiag) ? diagval : 0.f);
    sum += v * v;
  }
  red[tid] = sum;
  __syncthreads();
  for (int s = 128; s > 0; s >>= 1) { if (tid < s) red[tid] += red[tid + s]; __syncthreads(); }
  if (tid == 0) out[b] = sqrtf(red[0]);
}

// trace (pre-mask) -> num[b]; zero diag; symmetric degree normalize
__global__ void k_oadjnorm(float* __restrict__ oadj, float* __restrict__ num) {
  __shared__ float d[K2c];
  const int b = blockIdx.x, i = threadIdx.x;
  float* A = oadj + b * K2c * K2c;
  if (i == 0) {
    float tr = 0.f;
    for (int k = 0; k < K2c; ++k) tr += A[k * K2c + k];
    num[b] = tr;
  }
  __syncthreads();
  A[i * K2c + i] = 0.f;
  __syncthreads();
  float r = 0.f;
  for (int j = 0; j < K2c; ++j) r += A[i * K2c + j];
  d[i] = sqrtf(r) + EPSc;
  __syncthreads();
  for (int j = 0; j < K2c; ++j) A[i * K2c + j] /= (d[i] * d[j]);
}

// h3 = (oadj @ outp) @ rel_w + rel_b + outp @ root_w   (tiny, scalar)
__global__ void k_conv2(const float* __restrict__ oadj, const float* __restrict__ outp,
                        const float* __restrict__ relw, const float* __restrict__ relb,
                        const float* __restrict__ rootw, float* __restrict__ h3) {
  int idx = blockIdx.x * blockDim.x + threadIdx.x;
  if (idx >= Bc * K2c * HIDc) return;
  int b = idx / (K2c * HIDc), rem = idx % (K2c * HIDc), i = rem / HIDc, c = rem % HIDc;
  const float* A = oadj + b * K2c * K2c;
  const float* X = outp + b * K2c * HIDc;
  float acc = relb[c];
  for (int k = 0; k < HIDc; ++k) {
    float t = 0.f;
    for (int j = 0; j < K2c; ++j) t += A[i * K2c + j] * X[j * HIDc + k];
    acc += t * relw[k * HIDc + c];
  }
  for (int k = 0; k < HIDc; ++k) acc += X[i * HIDc + k] * rootw[k * HIDc + c];
  h3[idx] = acc;
}

__global__ void k_hsum(const float* __restrict__ h3, float* __restrict__ hs) {
  int idx = blockIdx.x * blockDim.x + threadIdx.x;
  if (idx >= Bc * HIDc) return;
  int b = idx / HIDc, c = idx % HIDc;
  float s = 0.f;
  for (int i = 0; i < K2c; ++i) s += h3[(b * K2c + i) * HIDc + c];
  hs[idx] = s;
}

// final head: lin2+relu, lin3, log_softmax, plus the two scalar losses
__global__ __launch_bounds__(32) void k_head(const float* __restrict__ hs,
                                             const float* __restrict__ l2w,
                                             const float* __restrict__ l2b,
                                             const float* __restrict__ l3w,
                                             const float* __restrict__ l3b,
                                             const float* __restrict__ scal,
                                             float* __restrict__ out) {
  const int t = threadIdx.x;
  if (t < Bc) {
    float hr[HIDc];
    for (int c = 0; c < HIDc; ++c) {
      float a = l2b[c];
      for (int k = 0; k < HIDc; ++k) a += hs[t * HIDc + k] * l2w[k * HIDc + c];
      hr[c] = fmaxf(a, 0.f);
    }
    float lg[COUTc], mx = -1e30f;
    for (int c = 0; c < COUTc; ++c) {
      float a = l3b[c];
      for (int k = 0; k < HIDc; ++k) a += hr[k] * l3w[k * COUTc + c];
      lg[c] = a;
      mx = fmaxf(mx, a);
    }
    float s = 0.f;
    for (int c = 0; c < COUTc; ++c) s += expf(lg[c] - mx);
    float lse = mx + logf(s);
    for (int c = 0; c < COUTc; ++c) out[t * COUTc + c] = lg[c] - lse;
  } else if (t == 16) {
    float ct = 0.f, o1 = 0.f;
    for (int b = 0; b < Bc; ++b) {
      ct += (scal[SC_SDS + b] - scal[SC_SAS + b]) / (scal[SC_SDS + b] + EPSc);
      o1 += scal[SC_O1 + b];
    }
    out[Bc * COUTc] = ct / Bc + o1 / Bc;
  } else if (t == 17) {
    float mc = 0.f, o2 = 0.f;
    for (int b = 0; b < Bc; ++b) {
      mc += scal[SC_NUM + b] / (scal[SC_DEN + b] + EPSc);
      o2 += scal[SC_O2 + b];
    }
    out[Bc * COUTc + 1] = -mc / Bc + o2 / Bc;
  }
}

// ---------------------------- launcher ----------------------------
extern "C" void kernel_launch(void* const* d_in, const int* in_sizes, int n_in,
                              void* d_out, int out_size, void* d_ws, size_t ws_size,
                              hipStream_t stream) {
  const float* x      = (const float*)d_in[0];
  const float* lin1_w = (const float*)d_in[1];
  const float* lin1_b = (const float*)d_in[2];
  const float* pool1_w = (const float*)d_in[3];
  const float* pool1_b = (const float*)d_in[4];
  const float* pool2_w = (const float*)d_in[5];
  const float* pool2_b = (const float*)d_in[6];
  const float* c1_rel_w = (const float*)d_in[7];
  const float* c1_rel_b = (const float*)d_in[8];
  const float* c1_root_w = (const float*)d_in[9];
  const float* c2_rel_w = (const float*)d_in[10];
  const float* c2_rel_b = (const float*)d_in[11];
  const float* c2_root_w = (const float*)d_in[12];
  const float* lin2_w = (const float*)d_in[13];
  const float* lin2_b = (const float*)d_in[14];
  const float* lin3_w = (const float*)d_in[15];
  const float* lin3_b = (const float*)d_in[16];
  const int* ei = (const int*)d_in[17];
  float* out = (float*)d_out;

  float* ws   = (float*)d_ws;
  float* agg  = ws + OFF_AGG;
  float* t2   = ws + OFF_T2;
  float* odeg = ws + OFF_ODEG;
  float* deg2 = ws + OFF_DEG2;
  float* degA2 = ws + OFF_DEGA2;
  float* scal = ws + OFF_SCAL;
  float* h1   = ws + OFF_H1;
  float* h2   = ws + OFF_H2;
  float* s1   = ws + OFF_S1;
  float* s2   = ws + OFF_S2;
  float* q1   = ws + OFF_Q1;
  float* q2   = ws + OFF_Q2;
  float* edot = ws + OFF_EDOT;
  float* ew   = ws + OFF_EW;
  float* ss1  = ws + OFF_SS1;
  float* ss2  = ws + OFF_SS2;
  float* outp = ws + OFF_OUTP;
  float* oadj = ws + OFF_OADJ;
  float* h3   = ws + OFF_H3;
  float* hs   = ws + OFF_HS;

  k_zero<<<(ZTOT + 255) / 256, 256, 0, stream>>>(ws, ZTOT);
  k_lin1<<<NBc / 16, 64, 0, stream>>>(x, lin1_w, lin1_b, h1);
  k_pool1<<<NBc / 16, 256, 0, stream>>>(h1, pool1_w, pool1_b, s1);
  k_softmax1<<<NBc / 256, 256, 0, stream>>>(s1, q1);
  k_edges1<<<Ec / 256, 256, 0, stream>>>(ei, s1, edot, odeg, scal + SC_SAS);
  k_sds<<<NBc / 256, 256, 0, stream>>>(odeg, q1, scal + SC_SDS);
  k_ew1<<<Ec / 256, 256, 0, stream>>>(ei, q1, edot, ew, deg2);
  k_ew2<<<Ec / 256, 256, 0, stream>>>(ei, deg2, ew, degA2);
  k_scatter_h<<<(Ec * HIDc) / 256, 256, 0, stream>>>(ei, ew, h1, agg);
  k_conv1<<<NBc / 16, 64, 0, stream>>>(agg, h1, c1_rel_w, c1_rel_b, c1_root_w, h2);
  k_pool2<<<NBc / 16, 32, 0, stream>>>(h2, pool2_w, pool2_b, s2);
  k_softmax2<<<NBc / 256, 256, 0, stream>>>(s2, q2);
  k_den2<<<NBc / 256, 256, 0, stream>>>(degA2, q2, scal + SC_DEN);
  k_scatter_t2<<<(Ec * K2c) / 256, 256, 0, stream>>>(ei, ew, s2, t2);
  k_ss1<<<Bc * 49, 32, 0, stream>>>(s1, ss1);
  k_bgemms<<<Bc, 64, 0, stream>>>(s2, h2, t2, outp, oadj, ss2);
  k_ortho<<<Bc, 256, 0, stream>>>(ss1, K1P, K1c, 1.0f / sqrtf((float)K1c), scal + SC_O1);
  k_ortho<<<Bc, 256, 0, stream>>>(ss2, K2c, K2c, 1.0f / sqrtf((float)K2c), scal + SC_O2);
  k_oadjnorm<<<Bc, K2c, 0, stream>>>(oadj, scal + SC_NUM);
  k_conv2<<<(Bc * K2c * HIDc) / 256, 256, 0, stream>>>(oadj, outp, c2_rel_w, c2_rel_b,
                                                       c2_root_w, h3);
  k_hsum<<<(Bc * HIDc + 255) / 256, 256, 0, stream>>>(h3, hs);
  k_head<<<1, 32, 0, stream>>>(hs, lin2_w, lin2_b, lin3_w, lin3_b, scal, out);
  (void)in_sizes; (void)n_in; (void)out_size; (void)ws_size;
}
